// LSTM_22368189677938
// MI455X (gfx1250) — matrix operand order repaired and checked by reading, not compiled, not run on hardware
//
#include <hip/hip_runtime.h>

typedef __attribute__((ext_vector_type(16))) _Float16 v16h;
typedef __attribute__((ext_vector_type(8)))  _Float16 v8h;
typedef __attribute__((ext_vector_type(8)))  float    v8f;
typedef __attribute__((ext_vector_type(2)))  float    v2f;

#define H        64
#define TSEQ     2048
#define BATCH    512
#define MTILE    16      // batch rows per workgroup
#define NTHREADS 512     // 16 wave32s
#define GSTRIDE  260     // padded gate-row stride (256 cols + 4 pad -> conflict-free)

// Fast activations: single TRANS32 ops instead of IEEE divide sequences.
__device__ __forceinline__ float fast_sigmoid(float x) {
    return __builtin_amdgcn_rcpf(1.0f + __expf(-x));
}

__device__ __forceinline__ float fast_tanh(float x) {
#if __has_builtin(__builtin_amdgcn_tanhf)
    return __builtin_amdgcn_tanhf(x);          // native V_TANH_F32 on gfx1250
#else
    float e = __expf(2.0f * x);                // saturates correctly at +/-inf
    return 1.0f - 2.0f * __builtin_amdgcn_rcpf(e + 1.0f);
#endif
}

// Load a 16x32 f16 A-fragment from row-major half LDS [16][H].
// Lane L: M = L%16; K-halves: hi 0..7 -> 32*kb + 8*(L/16) + hi,
//                             hi 8..15 -> +16.  (ISA 7.12.2, 16-bit A 16x32)
__device__ __forceinline__ v16h load_hA(const _Float16* hl, int m, int g, int kb) {
    const int k0 = 32 * kb + 8 * g;
    v8h lo = *(const v8h*)(hl + m * H + k0);
    v8h hi = *(const v8h*)(hl + m * H + k0 + 16);
    v16h a;
#pragma unroll
    for (int i = 0; i < 8; ++i) { a[i] = lo[i]; a[i + 8] = hi[i]; }
    return a;
}

__global__ __launch_bounds__(NTHREADS)
void lstm2_fused_kernel(const float* __restrict__ x,
                        const float* __restrict__ wih0, const float* __restrict__ whh0,
                        const float* __restrict__ bih0, const float* __restrict__ bhh0,
                        const float* __restrict__ wih1, const float* __restrict__ whh1,
                        const float* __restrict__ bih1, const float* __restrict__ bhh1,
                        float* __restrict__ out)
{
    __shared__ __align__(16) float    gate_lds[16 * GSTRIDE];
    __shared__ __align__(16) _Float16 h0_lds[16 * H];
    __shared__ __align__(16) _Float16 h1_lds[16 * H];

    const int tid  = threadIdx.x;
    const int wave = tid >> 5;         // 0..15 : N-tile id
    const int lane = tid & 31;
    const int g    = lane >> 4;        // half-wave group
    const int ln   = lane & 15;        // M (A-frag) / N (B-frag,C)
    const int n0   = wave << 4;        // tile column base
    const int j    = n0 + ln;          // owned gate column 0..255
    const int base = blockIdx.x * MTILE;
    // wave-uniform gate id as an SGPR -> scalar branch (no exec save/restore)
    const int gate_u = __builtin_amdgcn_readfirstlane(wave >> 2); // 0:i 1:f 2:g 3:o

    // ---------------- one-time weight B-fragment preload ----------------
    // f32 B-frag of w_ih_l0^T (4x16): element v -> K = 2*g + v, N = ln
    v2f wih0B;
    wih0B.x = wih0[j * 4 + 2 * g + 0];
    wih0B.y = wih0[j * 4 + 2 * g + 1];

    // f16 B-frags (32x16 each): half hi -> K = 32*kb + 8*g + (hi&7) + 16*(hi>>3), N = ln
    v16h whh0B[2], wih1B[2], whh1B[2];
#pragma unroll
    for (int kb = 0; kb < 2; ++kb) {
#pragma unroll
        for (int hi = 0; hi < 16; ++hi) {
            const int k = 32 * kb + 8 * g + (hi & 7) + ((hi >> 3) << 4);
            whh0B[kb][hi] = (_Float16)whh0[j * H + k];
            wih1B[kb][hi] = (_Float16)wih1[j * H + k];
            whh1B[kb][hi] = (_Float16)whh1[j * H + k];
        }
    }
    const float bias0 = bih0[j] + bhh0[j];
    const float bias1 = bih1[j] + bhh1[j];

    // ---------------- per-thread cell-state ownership ----------------
    const int scol  = tid & (H - 1);   // 0..63
    const int srow0 = tid >> 6;        // 0..7
    const int srow1 = srow0 + 8;
    float c00 = 0.f, c01 = 0.f, c10 = 0.f, c11 = 0.f;
    float h00 = 0.f, h01 = 0.f, h10 = 0.f, h11 = 0.f;

    // zero-init h buffers (h_{-1} = 0)
    h0_lds[tid]       = (_Float16)0.f;
    h0_lds[tid + 512] = (_Float16)0.f;
    h1_lds[tid]       = (_Float16)0.f;
    h1_lds[tid + 512] = (_Float16)0.f;
    __syncthreads();

    const float* xrow = x + ((size_t)(base + ln) * TSEQ) * 4 + 2 * g;

    // software-pipelined x fragment: prefetch t=0 before the loop
    v2f xA;
    xA.x = xrow[0];
    xA.y = xrow[1];

#pragma unroll 1
    for (int t = 0; t < TSEQ; ++t) {
        // issue prefetch of x[t+1] immediately; latency hides under the
        // whole iteration (barriers + 7 WMMAs) instead of stalling phase 1
        const int tn = (t + 1 < TSEQ) ? (t + 1) : t;
        v2f xA_next;
        xA_next.x = xrow[(size_t)tn * 4 + 0];
        xA_next.y = xrow[(size_t)tn * 4 + 1];

        // ---------- phase 1: layer-0 gate pre-activations ----------
        v8f acc;
#pragma unroll
        for (int r = 0; r < 8; ++r) acc[r] = bias0;

        acc = __builtin_amdgcn_wmma_f32_16x16x4_f32(false, xA, false, wih0B,
                                                    (short)0, acc, false, false);

        v16h a0 = load_hA(h0_lds, ln, g, 0);
        v16h a1 = load_hA(h0_lds, ln, g, 1);
        acc = __builtin_amdgcn_wmma_f32_16x16x32_f16(false, a0, false, whh0B[0],
                                                     (short)0, acc, false, false);
        acc = __builtin_amdgcn_wmma_f32_16x16x32_f16(false, a1, false, whh0B[1],
                                                     (short)0, acc, false, false);
        // activation (scalar-uniform branch) + stage gates to LDS
        if (gate_u == 2) {
#pragma unroll
            for (int r = 0; r < 8; ++r) acc[r] = fast_tanh(acc[r]);
        } else {
#pragma unroll
            for (int r = 0; r < 8; ++r) acc[r] = fast_sigmoid(acc[r]);
        }
#pragma unroll
        for (int r = 0; r < 8; ++r)
            gate_lds[(8 * g + r) * GSTRIDE + j] = acc[r];
        __syncthreads();

        // ---------- phase 2: layer-0 cell/hidden update ----------
        {
            const float* gl0 = gate_lds + srow0 * GSTRIDE + scol;
            const float* gl1 = gate_lds + srow1 * GSTRIDE + scol;
            float i0 = gl0[0], f0 = gl0[64], gg0 = gl0[128], o0 = gl0[192];
            float i1 = gl1[0], f1 = gl1[64], gg1 = gl1[128], o1 = gl1[192];
            c00 = f0 * c00 + i0 * gg0;  h00 = o0 * fast_tanh(c00);
            c01 = f1 * c01 + i1 * gg1;  h01 = o1 * fast_tanh(c01);
            h0_lds[srow0 * H + scol] = (_Float16)h00;
            h0_lds[srow1 * H + scol] = (_Float16)h01;
        }
        __syncthreads();

        // ---------- phase 3: layer-1 gate pre-activations ----------
        v8f acc1;
#pragma unroll
        for (int r = 0; r < 8; ++r) acc1[r] = bias1;

        v16h b0 = load_hA(h0_lds, ln, g, 0);   // input = h0_t
        v16h b1 = load_hA(h0_lds, ln, g, 1);
        acc1 = __builtin_amdgcn_wmma_f32_16x16x32_f16(false, b0, false, wih1B[0],
                                                      (short)0, acc1, false, false);
        acc1 = __builtin_amdgcn_wmma_f32_16x16x32_f16(false, b1, false, wih1B[1],
                                                      (short)0, acc1, false, false);
        v16h r0 = load_hA(h1_lds, ln, g, 0);   // recurrent = h1_{t-1}
        v16h r1 = load_hA(h1_lds, ln, g, 1);
        acc1 = __builtin_amdgcn_wmma_f32_16x16x32_f16(false, r0, false, whh1B[0],
                                                      (short)0, acc1, false, false);
        acc1 = __builtin_amdgcn_wmma_f32_16x16x32_f16(false, r1, false, whh1B[1],
                                                      (short)0, acc1, false, false);
        if (gate_u == 2) {
#pragma unroll
            for (int r = 0; r < 8; ++r) acc1[r] = fast_tanh(acc1[r]);
        } else {
#pragma unroll
            for (int r = 0; r < 8; ++r) acc1[r] = fast_sigmoid(acc1[r]);
        }
#pragma unroll
        for (int r = 0; r < 8; ++r)
            gate_lds[(8 * g + r) * GSTRIDE + j] = acc1[r];
        __syncthreads();

        // ---------- phase 4: layer-1 update + output write ----------
        {
            const float* gl0 = gate_lds + srow0 * GSTRIDE + scol;
            const float* gl1 = gate_lds + srow1 * GSTRIDE + scol;
            float i0 = gl0[0], f0 = gl0[64], gg0 = gl0[128], o0 = gl0[192];
            float i1 = gl1[0], f1 = gl1[64], gg1 = gl1[128], o1 = gl1[192];
            c10 = f0 * c10 + i0 * gg0;  h10 = o0 * fast_tanh(c10);
            c11 = f1 * c11 + i1 * gg1;  h11 = o1 * fast_tanh(c11);
            h1_lds[srow0 * H + scol] = (_Float16)h10;
            h1_lds[srow1 * H + scol] = (_Float16)h11;
            out[((size_t)(base + srow0) * TSEQ + t) * H + scol] = h10;
            out[((size_t)(base + srow1) * TSEQ + t) * H + scol] = h11;
        }
        __syncthreads();

        xA = xA_next;   // rotate software pipeline
    }

    // ---------------- final states: h_n [2,B,H], c_n [2,B,H] ----------------
    const size_t OUT1 = (size_t)BATCH * TSEQ * H;
    float* hn = out + OUT1;
    float* cn = out + OUT1 + (size_t)2 * BATCH * H;
    hn[(size_t)(0 * BATCH + base + srow0) * H + scol] = h00;
    hn[(size_t)(0 * BATCH + base + srow1) * H + scol] = h01;
    hn[(size_t)(1 * BATCH + base + srow0) * H + scol] = h10;
    hn[(size_t)(1 * BATCH + base + srow1) * H + scol] = h11;
    cn[(size_t)(0 * BATCH + base + srow0) * H + scol] = c00;
    cn[(size_t)(0 * BATCH + base + srow1) * H + scol] = c01;
    cn[(size_t)(1 * BATCH + base + srow0) * H + scol] = c10;
    cn[(size_t)(1 * BATCH + base + srow1) * H + scol] = c11;
}

extern "C" void kernel_launch(void* const* d_in, const int* in_sizes, int n_in,
                              void* d_out, int out_size, void* d_ws, size_t ws_size,
                              hipStream_t stream) {
    const float* x    = (const float*)d_in[0];
    const float* wih0 = (const float*)d_in[1];
    const float* whh0 = (const float*)d_in[2];
    const float* bih0 = (const float*)d_in[3];
    const float* bhh0 = (const float*)d_in[4];
    const float* wih1 = (const float*)d_in[5];
    const float* whh1 = (const float*)d_in[6];
    const float* bih1 = (const float*)d_in[7];
    const float* bhh1 = (const float*)d_in[8];
    float* out = (float*)d_out;

    dim3 grid(BATCH / MTILE);   // 32 workgroups
    dim3 block(NTHREADS);       // 16 wave32s
    lstm2_fused_kernel<<<grid, block, 0, stream>>>(x, wih0, whh0, bih0, bhh0,
                                                   wih1, whh1, bih1, bhh1, out);
}
